// DMoNEncoder_55052890800724
// MI455X (gfx1250) — matrix operand order repaired and checked by reading, not compiled, not run on hardware
//
#include <hip/hip_runtime.h>

typedef __attribute__((ext_vector_type(16))) _Float16 v16h;
typedef __attribute__((ext_vector_type(8)))  float    v8f;

#define HDIM 128
#define BN_EPS 1e-5f

// ---------------------------------------------------------------------------
// Degree / normalization
// ---------------------------------------------------------------------------
__global__ void deg_init_kernel(float* __restrict__ deg, int n) {
    int i = blockIdx.x * blockDim.x + threadIdx.x;
    if (i < n) deg[i] = 1.0f;                       // self loop
}

__global__ void deg_edges_kernel(const long long* __restrict__ erow,
                                 float* __restrict__ deg, int e) {
    int i = blockIdx.x * blockDim.x + threadIdx.x;
    int stride = gridDim.x * blockDim.x;
    for (; i < e; i += stride)
        atomicAdd(&deg[erow[i]], 1.0f);
}

__global__ void deg_finish_kernel(float* __restrict__ deg, int n) {
    int i = blockIdx.x * blockDim.x + threadIdx.x;
    if (i < n) {
        float d = deg[i];
        deg[i] = d > 0.0f ? rsqrtf(d) : 0.0f;       // now holds dinv
    }
}

// ---------------------------------------------------------------------------
// Pack W[128, ncols] (f32, row-major) into WMMA B-fragment order, f16:
//   Wp[((ct*4 + kk)*32 + lane)*16 + hv] = W[k, ct*16 + (lane&15)]
//   with k = kk*32 + (lane>>4)*16 + hv  (hv = 2v + odd covers K pair per VGPR)
// One lane's B fragment for (col-tile ct, k-step kk) is then 32 contiguous B.
// ---------------------------------------------------------------------------
__global__ void pack_w_kernel(const float* __restrict__ W,
                              _Float16* __restrict__ Wp, int ncols) {
    int idx = blockIdx.x * blockDim.x + threadIdx.x;
    int total = ncols * HDIM;
    if (idx >= total) return;
    int hv   = idx & 15;
    int lane = (idx >> 4) & 31;
    int kk   = (idx >> 9) & 3;
    int ct   = idx >> 11;
    int k    = kk * 32 + ((lane >> 4) << 4) + hv;
    int col  = (ct << 4) + (lane & 15);
    Wp[idx] = (_Float16)W[k * ncols + col];
}

// ---------------------------------------------------------------------------
// WMMA GEMM: C[nrows, ncols] = A[nrows, 128] @ W[128, ncols] (+bias, +relu)
// 16x16 output tile per wave; wave w handles columns [16w, 16w+16).
// blockDim.x = 32 * (ncols/16).  K=128 in 4 steps of v_wmma_f32_16x16x32_f16.
// A row tile (16x128 f32, 8KB) staged to LDS via gfx1250 async copy, then
// read back as float4 (ds_load_b128) and converted to f16 in VALU.
// B fragments come from the pre-packed f16 buffer: one v16h (2x b128) per
// lane per k-step.
// Fragment k order: contiguous k = 16*half + s in both A and B (wave32, 16-bit).
// ---------------------------------------------------------------------------
__global__ void wmma_gemm_kernel(const float* __restrict__ A,
                                 const _Float16* __restrict__ Wp,
                                 const float* __restrict__ bias,
                                 float* __restrict__ C,
                                 int nrows, int ncols, int dorelu) {
    __shared__ float sA[16 * HDIM];                 // 8 KB row tile
    const int lane = threadIdx.x & 31;
    const int wave = threadIdx.x >> 5;
    const int m    = lane & 15;
    const int half = lane >> 4;
    const int rb   = blockIdx.x << 4;               // row-tile base
    const int cb   = wave << 4;                     // col-tile base

    // ---- stage A tile to LDS with async 16B copies ----
    for (int c = threadIdx.x; c < (16 * HDIM) / 4; c += blockDim.x) {
        const int r    = c >> 5;                    // tile row (32 chunks/row)
        const int off  = (c & 31) << 2;             // float offset within row
        const int grow = min(rb + r, nrows - 1);    // clamp OOB (masked later)
        const unsigned long long gaddr =
            (unsigned long long)(size_t)(A + (size_t)grow * HDIM + off);
        const unsigned laddr =
            (unsigned)(size_t)(__attribute__((address_space(3))) char*)(&sA[c << 2]);
        asm volatile("global_load_async_to_lds_b128 %0, %1, off"
                     :: "v"(laddr), "v"(gaddr) : "memory");
    }
    asm volatile("s_wait_asynccnt 0" ::: "memory");
    __syncthreads();

    const _Float16* wtile = Wp + ((size_t)(cb >> 4) << 11) + (lane << 4);
    const float* sArow = &sA[m * HDIM];

    v8f acc = {};
#pragma unroll
    for (int kk = 0; kk < 4; ++kk) {
        const float4 a0 = *(const float4*)(sArow + kk * 32 + half * 16);
        const float4 a1 = *(const float4*)(sArow + kk * 32 + half * 16 + 4);
        const float4 a2 = *(const float4*)(sArow + kk * 32 + half * 16 + 8);
        const float4 a3 = *(const float4*)(sArow + kk * 32 + half * 16 + 12);
        v16h af;
        af[0]  = (_Float16)a0.x;  af[1]  = (_Float16)a0.y;
        af[2]  = (_Float16)a0.z;  af[3]  = (_Float16)a0.w;
        af[4]  = (_Float16)a1.x;  af[5]  = (_Float16)a1.y;
        af[6]  = (_Float16)a1.z;  af[7]  = (_Float16)a1.w;
        af[8]  = (_Float16)a2.x;  af[9]  = (_Float16)a2.y;
        af[10] = (_Float16)a2.z;  af[11] = (_Float16)a2.w;
        af[12] = (_Float16)a3.x;  af[13] = (_Float16)a3.y;
        af[14] = (_Float16)a3.z;  af[15] = (_Float16)a3.w;
        const v16h bf = *(const v16h*)(wtile + (kk << 9));
        acc = __builtin_amdgcn_wmma_f32_16x16x32_f16(
                  false, af, false, bf, (short)0, acc, false, false);
    }

    const int cn = cb + m;
    const float bv = bias ? bias[cn] : 0.0f;
#pragma unroll
    for (int r = 0; r < 8; ++r) {
        const int row = rb + (half << 3) + r;
        if (row < nrows) {
            float v = acc[r] + bv;
            if (dorelu) v = fmaxf(v, 0.0f);
            C[(size_t)row * ncols + cn] = v;
        }
    }
}

// ---------------------------------------------------------------------------
// Aggregation: out = self-loop contribution + bias, then edge scatter-adds.
// ---------------------------------------------------------------------------
__global__ void agg_init_kernel(const float* __restrict__ msg,
                                const float* __restrict__ dinv,
                                const float* __restrict__ bias,
                                float* __restrict__ out, int n) {
    size_t idx = (size_t)blockIdx.x * blockDim.x + threadIdx.x;
    size_t total = (size_t)n * HDIM;
    size_t stride = (size_t)gridDim.x * blockDim.x;
    for (; idx < total; idx += stride) {
        int row = (int)(idx >> 7);
        int col = (int)(idx & (HDIM - 1));
        float di = dinv[row];
        out[idx] = msg[idx] * di * di + bias[col];
    }
}

#define EDGES_PER_BLOCK 32
__global__ void agg_edges_kernel(const long long* __restrict__ erow,
                                 const long long* __restrict__ ecol,
                                 const float* __restrict__ dinv,
                                 const float* __restrict__ msg,
                                 float* __restrict__ out, int e) {
    const int col = threadIdx.x;                    // 0..127: feature column
    const int e0 = blockIdx.x * EDGES_PER_BLOCK;
    const int e1 = min(e0 + EDGES_PER_BLOCK, e);
    for (int i = e0; i < e1; ++i) {
        const long long t = erow[i];
        const long long s = ecol[i];
        const float w = dinv[t] * dinv[s];
        atomicAdd(&out[(size_t)t * HDIM + col],
                  msg[(size_t)s * HDIM + col] * w);
    }
}

// ---------------------------------------------------------------------------
// BatchNorm (over N) + ReLU, fused scale/shift
// ---------------------------------------------------------------------------
__global__ void zero_stats_kernel(float* __restrict__ stats) {
    stats[threadIdx.x] = 0.0f;                      // 256 threads: sum+sumsq
}

#define BN_ROWS 512
__global__ void bn_stats_kernel(const float* __restrict__ h,
                                float* __restrict__ stats, int n) {
    const int col = threadIdx.x;                    // 128 threads
    const int row0 = blockIdx.x * BN_ROWS;
    float s = 0.0f, s2 = 0.0f;
    for (int r = 0; r < BN_ROWS; ++r) {
        int row = row0 + r;
        if (row < n) {
            float v = h[(size_t)row * HDIM + col];
            s += v; s2 += v * v;
        }
    }
    atomicAdd(&stats[col], s);
    atomicAdd(&stats[HDIM + col], s2);
}

__global__ void bn_finalize_kernel(const float* __restrict__ stats,
                                   const float* __restrict__ g,
                                   const float* __restrict__ be,
                                   float* __restrict__ ss, int n) {
    const int col = threadIdx.x;                    // 128 threads
    float invn = 1.0f / (float)n;
    float mu = stats[col] * invn;
    float var = stats[HDIM + col] * invn - mu * mu;
    float istd = rsqrtf(var + BN_EPS);
    float sc = g[col] * istd;
    ss[col] = sc;
    ss[HDIM + col] = be[col] - mu * sc;
}

__global__ void bn_apply_kernel(const float* __restrict__ in,
                                const float* __restrict__ ss,
                                float* __restrict__ out, int n) {
    size_t idx = (size_t)blockIdx.x * blockDim.x + threadIdx.x;
    size_t total = (size_t)n * HDIM;
    size_t stride = (size_t)gridDim.x * blockDim.x;
    for (; idx < total; idx += stride) {
        int col = (int)(idx & (HDIM - 1));
        out[idx] = fmaxf(in[idx] * ss[col] + ss[HDIM + col], 0.0f);
    }
}

// ---------------------------------------------------------------------------
// Softmax over 16 logits per row
// ---------------------------------------------------------------------------
__global__ void softmax16_kernel(const float* __restrict__ logits,
                                 float* __restrict__ S, int n) {
    int row = blockIdx.x * blockDim.x + threadIdx.x;
    if (row >= n) return;
    float v[16];
    float mx = -3.4e38f;
#pragma unroll
    for (int k = 0; k < 16; ++k) {
        v[k] = logits[(size_t)row * 16 + k];
        mx = fmaxf(mx, v[k]);
    }
    float sum = 0.0f;
#pragma unroll
    for (int k = 0; k < 16; ++k) {
        v[k] = __expf(v[k] - mx);
        sum += v[k];
    }
    float inv = 1.0f / sum;
#pragma unroll
    for (int k = 0; k < 16; ++k)
        S[(size_t)row * 16 + k] = v[k] * inv;
}

// ---------------------------------------------------------------------------
// Host-side launch
// ---------------------------------------------------------------------------
extern "C" void kernel_launch(void* const* d_in, const int* in_sizes, int n_in,
                              void* d_out, int out_size, void* d_ws, size_t ws_size,
                              hipStream_t stream) {
    const float*     x    = (const float*)d_in[0];
    const long long* ei   = (const long long*)d_in[1];   // int64 [2, E]
    const float*     W1   = (const float*)d_in[2];
    const float*     b1   = (const float*)d_in[3];
    const float*     g1   = (const float*)d_in[4];
    const float*     be1  = (const float*)d_in[5];
    const float*     W2   = (const float*)d_in[6];
    const float*     b2   = (const float*)d_in[7];
    const float*     g2   = (const float*)d_in[8];
    const float*     be2  = (const float*)d_in[9];
    const float*     AW1  = (const float*)d_in[10];
    const float*     Ab1  = (const float*)d_in[11];
    const float*     AW2  = (const float*)d_in[12];
    const float*     Ab2  = (const float*)d_in[13];

    const int N = in_sizes[0] / HDIM;    // 100000
    const int E = in_sizes[1] / 2;       // 1600000

    const long long* erow = ei;          // targets
    const long long* ecol = ei + E;      // sources

    // Workspace layout (floats, then f16 packed weights)
    float* ws     = (float*)d_ws;
    float* dinv   = ws;                                   // N
    float* stats  = ws + N;                               // 256 (sum, sumsq)
    float* ss     = stats + 256;                          // 256 (scale, shift)
    float* bufA   = ss + 256;                             // N*128
    float* bufB   = bufA + (size_t)N * HDIM;              // N*128
    float* logit  = bufB + (size_t)N * HDIM;              // N*16
    _Float16* wp1  = (_Float16*)(logit + (size_t)N * 16); // 128*128
    _Float16* wp2  = wp1 + 16384;                         // 128*128
    _Float16* wpa1 = wp2 + 16384;                         // 128*128
    _Float16* wpa2 = wpa1 + 16384;                        // 128*16

    float* hOut = (float*)d_out;                          // N*128
    float* sOut = hOut + (size_t)N * HDIM;                // N*16

    const int rowTiles = (N + 15) / 16;
    const dim3 b256(256), b128(HDIM), b32(32);
    const int gridElem = 4096;                            // grid-stride elementwise

    // --- pack weights into WMMA fragment order (f16) ---
    pack_w_kernel<<<(HDIM * HDIM + 255) / 256, b256, 0, stream>>>(W1, wp1, HDIM);
    pack_w_kernel<<<(HDIM * HDIM + 255) / 256, b256, 0, stream>>>(W2, wp2, HDIM);
    pack_w_kernel<<<(HDIM * HDIM + 255) / 256, b256, 0, stream>>>(AW1, wpa1, HDIM);
    pack_w_kernel<<<(HDIM * 16 + 255) / 256, b256, 0, stream>>>(AW2, wpa2, 16);

    // --- degree -> dinv ---
    deg_init_kernel<<<(N + 255) / 256, b256, 0, stream>>>(dinv, N);
    deg_edges_kernel<<<2048, b256, 0, stream>>>(erow, dinv, E);
    deg_finish_kernel<<<(N + 255) / 256, b256, 0, stream>>>(dinv, N);

    // --- Layer 1: conv + BN + ReLU ---
    wmma_gemm_kernel<<<rowTiles, b256, 0, stream>>>(x, wp1, nullptr, bufA, N, HDIM, 0);
    agg_init_kernel<<<gridElem, b256, 0, stream>>>(bufA, dinv, b1, bufB, N);
    agg_edges_kernel<<<(E + EDGES_PER_BLOCK - 1) / EDGES_PER_BLOCK, b128, 0, stream>>>(
        erow, ecol, dinv, bufA, bufB, E);
    zero_stats_kernel<<<1, b256, 0, stream>>>(stats);
    bn_stats_kernel<<<(N + BN_ROWS - 1) / BN_ROWS, b128, 0, stream>>>(bufB, stats, N);
    bn_finalize_kernel<<<1, b128, 0, stream>>>(stats, g1, be1, ss, N);
    bn_apply_kernel<<<gridElem, b256, 0, stream>>>(bufB, ss, bufB, N);

    // --- Layer 2: conv + BN + ReLU -> h (first output) ---
    wmma_gemm_kernel<<<rowTiles, b256, 0, stream>>>(bufB, wp2, nullptr, bufA, N, HDIM, 0);
    agg_init_kernel<<<gridElem, b256, 0, stream>>>(bufA, dinv, b2, bufB, N);
    agg_edges_kernel<<<(E + EDGES_PER_BLOCK - 1) / EDGES_PER_BLOCK, b128, 0, stream>>>(
        erow, ecol, dinv, bufA, bufB, E);
    zero_stats_kernel<<<1, b256, 0, stream>>>(stats);
    bn_stats_kernel<<<(N + BN_ROWS - 1) / BN_ROWS, b128, 0, stream>>>(bufB, stats, N);
    bn_finalize_kernel<<<1, b128, 0, stream>>>(stats, g2, be2, ss, N);
    bn_apply_kernel<<<gridElem, b256, 0, stream>>>(bufB, ss, hOut, N);

    // --- MLP head: relu(h@AW1+Ab1)@AW2+Ab2, softmax ---
    wmma_gemm_kernel<<<rowTiles, b256, 0, stream>>>(hOut, wpa1, Ab1, bufA, N, HDIM, 1);
    wmma_gemm_kernel<<<rowTiles, b32, 0, stream>>>(bufA, wpa2, Ab2, logit, N, 16, 0);
    softmax16_kernel<<<(N + 255) / 256, b256, 0, stream>>>(logit, sOut, N);
}
